// VisionTransformer_50242527428753
// MI455X (gfx1250) — matrix operand order repaired and checked by reading, not compiled, not run on hardware
//
#include <hip/hip_runtime.h>
#include <math.h>

// ---------------------------------------------------------------------------
// MoE Vision Transformer forward, MI455X (gfx1250), wave32 + bf16 WMMA.
// GEMM operands pre-converted to bf16 (weights transposed to W^T[n][k]);
// K-tiles double-buffered in LDS via GLOBAL_LOAD_ASYNC_TO_LDS_B128 with an
// in-order ASYNCcnt pipeline (s_wait_asynccnt 3) overlapping DMA with WMMA.
// ---------------------------------------------------------------------------

#define Dm   256
#define NHh  8
#define HDd  32
#define Ll   4
#define Ee   8
#define Bb   32
#define IMGs 224
#define NCc  1000
#define FFf  1024
#define Ss   197          // valid tokens
#define SP   224          // padded tokens (14*16, 7*32)
#define NPAT 196
#define MTOK (Bb * SP)    // 7168 padded token rows
#define MPAT (Bb * NPAT)  // 6272 patch rows

typedef unsigned short u16;
typedef __attribute__((ext_vector_type(16))) __bf16        v16bf;
typedef __attribute__((ext_vector_type(8)))  unsigned int  v8ui;
typedef __attribute__((ext_vector_type(8)))  float         v8f;

__device__ __forceinline__ u16 f2bf(float f) {
  unsigned u = __builtin_bit_cast(unsigned, f);
  unsigned r = u + 0x7FFFu + ((u >> 16) & 1u);  // round-to-nearest-even
  return (u16)(r >> 16);
}

__device__ __forceinline__ v8f wmma_bf16(v8ui a, v8ui b, v8f c) {
  return __builtin_amdgcn_wmma_f32_16x16x32_bf16(
      false, __builtin_bit_cast(v16bf, a),
      false, __builtin_bit_cast(v16bf, b),
      (short)0, c, false, false);
}

// LDS aperture truncates flat addresses to the wave-relative low 32 bits,
// which is exactly the VDST operand async-to-LDS expects.
__device__ __forceinline__ unsigned ldsOff(const void* p) {
  return (unsigned)(size_t)p;
}

// Async 16B copy global -> LDS (GV mode, tracked by ASYNCcnt).
__device__ __forceinline__ void async_cp16(unsigned lds_off, const void* g) {
  asm volatile("global_load_async_to_lds_b128 %0, %1, off"
               :: "v"(lds_off), "v"(g) : "memory");
}

__device__ __forceinline__ void async_wait0() {
  asm volatile("s_wait_asynccnt 0" ::: "memory");
}

// ASYNCcnt retires in order: <=3 outstanding means the 3 older (current-tile)
// copies are done while the 3 newer (next-tile) copies stream in behind.
__device__ __forceinline__ void async_wait3() {
  asm volatile("s_wait_asynccnt 3" ::: "memory");
}

// ---------------------------------------------------------------------------
// GEMM: C[M x Nn] = act(A[M x Kd] @ W[Kd x Nn] + bias (+ residual))
// A: bf16 row-major. WT: bf16 W^T, i.e. WT[n*Kd + k]. Requires M%128==0,
// Nn%64==0, Kd%32==0. Block: 256 thr = 8 waves; block tile 128x64; each wave
// computes 32x32 via 2x2 fragments -> 4 WMMA per K-step. Double-buffered
// async K pipeline. ACT: 0=none, 1=exact GELU. OUTBF: 0 -> f32 C, 1 -> bf16 C.
// ---------------------------------------------------------------------------
template <int ACT, int OUTBF>
__global__ __launch_bounds__(256) void k_gemm(
    const u16* __restrict__ A, const u16* __restrict__ WT,
    const float* __restrict__ bias, const float* __restrict__ res,
    void* __restrict__ Cout, int M, int Nn, int Kd)
{
  __shared__ u16 As[2][128][40];  // 128 rows x 32 k, stride 40 (80B: aligned, 20-bank)
  __shared__ u16 Bs[2][64][40];   // 64 n-rows x 32 k

  const int tid  = threadIdx.x;
  const int lane = tid & 31;
  const int wv   = tid >> 5;
  const int mw   = wv >> 1;        // 0..3  -> rows mw*32
  const int nw   = wv & 1;         // 0..1  -> cols nw*32
  const int brow = blockIdx.x * 128;
  const int bcol = blockIdx.y * 64;
  const int m    = lane & 15;
  const int hi   = lane >> 4;

  // per-thread staging assignments (fixed across K loop)
  const int ar   = tid >> 1;             // A row 0..127
  const int aseg = (tid & 1) * 16;       // A k-seg (16 elems = 32B)
  const int bn   = tid >> 2;             // B n-row 0..63
  const int bseg = (tid & 3) * 8;        // B k-seg (8 elems = 16B)
  const unsigned aoff0 = ldsOff(&As[0][ar][aseg]);
  const unsigned aoff1 = ldsOff(&As[1][ar][aseg]);
  const unsigned boff0 = ldsOff(&Bs[0][bn][bseg]);
  const unsigned boff1 = ldsOff(&Bs[1][bn][bseg]);
  const u16* agp = A  + (size_t)(brow + ar) * Kd + aseg;
  const u16* bgp = WT + (size_t)(bcol + bn) * Kd + bseg;

  v8f acc[2][2];
#pragma unroll
  for (int i = 0; i < 2; ++i)
#pragma unroll
    for (int j = 0; j < 2; ++j)
      acc[i][j] = v8f{0.f, 0.f, 0.f, 0.f, 0.f, 0.f, 0.f, 0.f};

  // prologue: stage K-tile 0 into buffer 0
  async_cp16(aoff0,      agp);
  async_cp16(aoff0 + 16, agp + 8);
  async_cp16(boff0,      bgp);

  int cb = 0;
  for (int k0 = 0; k0 < Kd; k0 += 32) {
    // issue next tile into the other buffer, then retire current tile's copies
    if (k0 + 32 < Kd) {
      const unsigned an = cb ? aoff0 : aoff1;
      const unsigned bnn = cb ? boff0 : boff1;
      async_cp16(an,      agp + k0 + 32);
      async_cp16(an + 16, agp + k0 + 40);
      async_cp16(bnn,     bgp + k0 + 32);
      async_wait3();
    } else {
      async_wait0();
    }
    __syncthreads();

    // fragments: contiguous (k,k+1) pairs -> ds_load_b32
    v8ui a0, a1, b0, b1;
#pragma unroll
    for (int p = 0; p < 8; ++p) {
      const int ka = 2 * p + 8 * hi + (p >= 4 ? 8 : 0);  // A 16x32 pattern
      a0[p] = *(const unsigned*)&As[cb][mw * 32 + m][ka];
      a1[p] = *(const unsigned*)&As[cb][mw * 32 + 16 + m][ka];
      const int kb = 2 * p + 8 * hi + (p >= 4 ? 8 : 0);  // B 32x16 pattern
      b0[p] = *(const unsigned*)&Bs[cb][nw * 32 + m][kb];
      b1[p] = *(const unsigned*)&Bs[cb][nw * 32 + 16 + m][kb];
    }
    acc[0][0] = wmma_bf16(a0, b0, acc[0][0]);
    acc[0][1] = wmma_bf16(a0, b1, acc[0][1]);
    acc[1][0] = wmma_bf16(a1, b0, acc[1][0]);
    acc[1][1] = wmma_bf16(a1, b1, acc[1][1]);
    __syncthreads();
    cb ^= 1;
  }

  // epilogue (C/D layout: lane col = lane&15, rows r + 8*(lane>>4))
#pragma unroll
  for (int j = 0; j < 2; ++j) {
    const int col = bcol + nw * 32 + j * 16 + m;
    const float bval = bias ? bias[col] : 0.f;
#pragma unroll
    for (int i = 0; i < 2; ++i) {
#pragma unroll
      for (int r = 0; r < 8; ++r) {
        const int row = brow + mw * 32 + i * 16 + r + 8 * hi;
        float v = acc[i][j][r] + bval;
        if (res) v += res[(size_t)row * Nn + col];
        if (ACT == 1) v = 0.5f * v * (1.0f + erff(v * 0.70710678118654752f));
        if (OUTBF) ((u16*)Cout)[(size_t)row * Nn + col] = f2bf(v);
        else       ((float*)Cout)[(size_t)row * Nn + col] = v;
      }
    }
  }
}

// ---------------------------------------------------------------------------
// Attention: one wave per (query-tile, head, batch). qkv is bf16,
// per token row (768): [q(256)|k(256)|v(256)], head h at +h*32. Output bf16.
// ---------------------------------------------------------------------------
__global__ __launch_bounds__(32) void k_attn(
    const u16* __restrict__ qkv, u16* __restrict__ o)
{
  const int qt = blockIdx.x;   // 0..13
  const int h  = blockIdx.y;   // 0..7
  const int b  = blockIdx.z;   // 0..31
  const int lane = threadIdx.x;
  const int m  = lane & 15;
  const int hi = lane >> 4;

  __shared__ float sc[16][SP];
  __shared__ u16   pb[16][SP + 8];

  const float  scale = 0.17677669529663687f;  // 1/sqrt(32)
  const size_t base  = (size_t)b * SP * 768;

  // Q fragment: pairs contiguous -> u32 global loads
  v8ui qf;
  {
    const u16* qp = qkv + base + (size_t)(qt * 16 + m) * 768 + h * HDd;
#pragma unroll
    for (int p = 0; p < 8; ++p) {
      const int ka = 2 * p + 8 * hi + (p >= 4 ? 8 : 0);
      qf[p] = *(const unsigned*)(qp + ka);
    }
  }

  // scores = Q @ K^T, 14 key tiles of 16
  for (int kt = 0; kt < SP / 16; ++kt) {
    v8ui kf;
    const u16* kp = qkv + base + (size_t)(kt * 16 + m) * 768 + Dm + h * HDd;
#pragma unroll
    for (int p = 0; p < 8; ++p) {
      const int kb = 2 * p + 8 * hi + (p >= 4 ? 8 : 0);   // B frag: k = head dim, n = key
      kf[p] = *(const unsigned*)(kp + kb);
    }
    v8f z = {0.f, 0.f, 0.f, 0.f, 0.f, 0.f, 0.f, 0.f};
    v8f s = wmma_bf16(qf, kf, z);
    const int col = kt * 16 + m;
#pragma unroll
    for (int r = 0; r < 8; ++r) {
      const float v = s[r] * scale;
      sc[r + 8 * hi][col] = (col < Ss) ? v : -3.0e38f;
    }
  }
  __syncthreads();

  // row softmax (lanes 0..15, one query row each)
  if (lane < 16) {
    float mx = -3.0e38f;
    for (int j = 0; j < Ss; ++j) mx = fmaxf(mx, sc[lane][j]);
    float sum = 0.f;
    for (int j = 0; j < Ss; ++j) {
      const float e = __expf(sc[lane][j] - mx);
      sc[lane][j] = e;
      sum += e;
    }
    const float inv = 1.f / sum;
    for (int j = 0; j < SP; ++j)
      pb[lane][j] = (j < Ss) ? f2bf(sc[lane][j] * inv) : (u16)0;
  }
  __syncthreads();

  // O = P @ V : 2 dim-tiles of 16, 7 key K-steps of 32
  for (int nt = 0; nt < 2; ++nt) {
    v8f acc = {0.f, 0.f, 0.f, 0.f, 0.f, 0.f, 0.f, 0.f};
    for (int kt = 0; kt < SP / 32; ++kt) {
      v8ui pf, vf;
#pragma unroll
      for (int p = 0; p < 8; ++p) {
        const int ka = 2 * p + 8 * hi + (p >= 4 ? 8 : 0);
        pf[p] = *(const unsigned*)&pb[m][kt * 32 + ka];
        const int kb = 2 * p + 8 * hi + (p >= 4 ? 8 : 0);   // key index within K-step
        const u16* vp = qkv + base + (size_t)(kt * 32 + kb) * 768 + 2 * Dm +
                        h * HDd + nt * 16 + m;
        vf[p] = (unsigned)vp[0] | ((unsigned)vp[768] << 16);
      }
      acc = wmma_bf16(pf, vf, acc);
    }
    const int dcol = h * HDd + nt * 16 + m;
#pragma unroll
    for (int r = 0; r < 8; ++r) {
      const int qrow = qt * 16 + r + 8 * hi;
      o[((size_t)b * SP + qrow) * Dm + dcol] = f2bf(acc[r]);
    }
  }
}

// ---------------------------------------------------------------------------
// LayerNorm over D=256, one wave per row. OUTBF: 1 -> bf16 out, 0 -> f32 out.
// ---------------------------------------------------------------------------
template <int OUTBF>
__global__ __launch_bounds__(256) void k_ln(
    const float* __restrict__ X, const float* __restrict__ g,
    const float* __restrict__ bta, void* __restrict__ Yout, int M)
{
  const int row  = blockIdx.x * 8 + (threadIdx.x >> 5);
  const int lane = threadIdx.x & 31;
  if (row >= M) return;
  const float* x = X + (size_t)row * Dm;
  float vals[8];
  float s = 0.f;
#pragma unroll
  for (int j = 0; j < 8; ++j) { vals[j] = x[lane + 32 * j]; s += vals[j]; }
#pragma unroll
  for (int off = 16; off; off >>= 1) s += __shfl_xor(s, off, 32);
  const float mean = s * (1.f / 256.f);
  float v = 0.f;
#pragma unroll
  for (int j = 0; j < 8; ++j) { const float d = vals[j] - mean; v += d * d; }
#pragma unroll
  for (int off = 16; off; off >>= 1) v += __shfl_xor(v, off, 32);
  const float inv = rsqrtf(v * (1.f / 256.f) + 1e-5f);
#pragma unroll
  for (int j = 0; j < 8; ++j) {
    const int d = lane + 32 * j;
    const float y = (vals[j] - mean) * inv * g[d] + bta[d];
    if (OUTBF) ((u16*)Yout)[(size_t)row * Dm + d] = f2bf(y);
    else       ((float*)Yout)[(size_t)row * Dm + d] = y;
  }
}

// ---------------------------------------------------------------------------
// Router: softmax(x @ rw + rb), top-2 renormalized -> dense weights[tok][E].
// ---------------------------------------------------------------------------
__global__ void k_router(const float* __restrict__ h, const float* __restrict__ rw,
                         const float* __restrict__ rb, float* __restrict__ wout, int M)
{
  const int t = blockIdx.x * blockDim.x + threadIdx.x;
  if (t >= M) return;
  const float* x = h + (size_t)t * Dm;
  float lg[Ee];
#pragma unroll
  for (int e = 0; e < Ee; ++e) lg[e] = rb[e];
  for (int d = 0; d < Dm; ++d) {
    const float xv = x[d];
#pragma unroll
    for (int e = 0; e < Ee; ++e) lg[e] += xv * rw[d * Ee + e];
  }
  float mx = lg[0];
#pragma unroll
  for (int e = 1; e < Ee; ++e) mx = fmaxf(mx, lg[e]);
  float sum = 0.f;
  float p[Ee];
#pragma unroll
  for (int e = 0; e < Ee; ++e) { p[e] = __expf(lg[e] - mx); sum += p[e]; }
  const float inv = 1.f / sum;
#pragma unroll
  for (int e = 0; e < Ee; ++e) p[e] *= inv;
  int i0 = 0;
#pragma unroll
  for (int e = 1; e < Ee; ++e) if (p[e] > p[i0]) i0 = e;
  int i1 = (i0 == 0) ? 1 : 0;
#pragma unroll
  for (int e = 0; e < Ee; ++e) if (e != i0 && p[e] > p[i1]) i1 = e;
  const float denom = 1.f / (p[i0] + p[i1]);
#pragma unroll
  for (int e = 0; e < Ee; ++e)
    wout[(size_t)t * Ee + e] =
        (e == i0) ? p[i0] * denom : ((e == i1) ? p[i1] * denom : 0.f);
}

// ---------------------------------------------------------------------------
// Elementwise / data-movement kernels
// ---------------------------------------------------------------------------
__global__ void k_zero(float* __restrict__ p, size_t n) {
  const size_t i = (size_t)blockIdx.x * blockDim.x + threadIdx.x;
  if (i < n) p[i] = 0.f;
}

__global__ void k_accum(float* __restrict__ nxt, const float* __restrict__ xe,
                        const float* __restrict__ w, int e, size_t n) {
  const size_t i = (size_t)blockIdx.x * blockDim.x + threadIdx.x;
  if (i < n) {
    const size_t t = i >> 8;  // / 256
    nxt[i] += w[t * Ee + e] * xe[i];
  }
}

// fp32 [nmat x Kd x Nn] row-major -> bf16 transposed [nmat x Nn x Kd]
__global__ void k_cvtT(const float* __restrict__ in, u16* __restrict__ out,
                       int Kd, int Nn, int nmat) {
  const size_t i = (size_t)blockIdx.x * blockDim.x + threadIdx.x;
  const size_t per = (size_t)Kd * Nn;
  if (i >= per * nmat) return;
  const size_t mat = i / per;
  const size_t rem = i - mat * per;
  const int k = (int)(rem / Nn);
  const int n = (int)(rem - (size_t)k * Nn);
  out[mat * per + (size_t)n * Kd + k] = f2bf(in[i]);
}

__global__ void k_im2col(const float* __restrict__ x, u16* __restrict__ A) {
  const size_t i = (size_t)blockIdx.x * blockDim.x + threadIdx.x;
  const size_t total = (size_t)MPAT * 768;
  if (i >= total) return;
  const int f = (int)(i % 768);
  const int t = (int)((i / 768) % NPAT);
  const int b = (int)(i / ((size_t)768 * NPAT));
  const int c = f >> 8, py = (f >> 4) & 15, px = f & 15;
  const int gy = t / 14, gx = t % 14;
  A[i] = f2bf(x[(((size_t)b * 3 + c) * IMGs + gy * 16 + py) * IMGs + gx * 16 + px]);
}

__global__ void k_embed(const float* __restrict__ Cpatch, const float* __restrict__ cls,
                        const float* __restrict__ pos, float* __restrict__ h) {
  const size_t i = (size_t)blockIdx.x * blockDim.x + threadIdx.x;
  const size_t total = (size_t)MTOK * Dm;
  if (i >= total) return;
  const int d = (int)(i & (Dm - 1));
  const int s = (int)((i >> 8) % SP);
  const int b = (int)(i / ((size_t)SP * Dm));
  float v;
  if (s == 0)         v = cls[d] + pos[d];
  else if (s <= NPAT) v = Cpatch[((size_t)b * NPAT + (s - 1)) * Dm + d] + pos[(size_t)s * Dm + d];
  else                v = 0.f;
  h[i] = v;
}

__global__ void k_head(const float* __restrict__ hln, const float* __restrict__ hw,
                       const float* __restrict__ hb, float* __restrict__ out) {
  const int i = blockIdx.x * blockDim.x + threadIdx.x;
  if (i >= Bb * NCc) return;
  const int b = i / NCc, n = i % NCc;
  const float* x = hln + (size_t)b * SP * Dm;  // cls row (s == 0)
  float s = hb[n];
  for (int d = 0; d < Dm; ++d) s += x[d] * hw[(size_t)d * NCc + n];
  out[i] = s;
}

// ---------------------------------------------------------------------------
// Host orchestration
// ---------------------------------------------------------------------------
extern "C" void kernel_launch(void* const* d_in, const int* in_sizes, int n_in,
                              void* d_out, int out_size, void* d_ws, size_t ws_size,
                              hipStream_t stream) {
  (void)in_sizes; (void)n_in; (void)out_size; (void)ws_size;
  const float* x        = (const float*)d_in[0];
  const float* patch_w  = (const float*)d_in[1];
  const float* patch_b  = (const float*)d_in[2];
  const float* cls_tok  = (const float*)d_in[3];
  const float* pos_emb  = (const float*)d_in[4];
  const float* router_w = (const float*)d_in[5];
  const float* router_b = (const float*)d_in[6];
  const float* ln1_g    = (const float*)d_in[7];
  const float* ln1_b    = (const float*)d_in[8];
  const float* qkv_w    = (const float*)d_in[9];
  const float* qkv_b    = (const float*)d_in[10];
  const float* proj_w   = (const float*)d_in[11];
  const float* proj_b   = (const float*)d_in[12];
  const float* ln2_g    = (const float*)d_in[13];
  const float* ln2_b    = (const float*)d_in[14];
  const float* fc1_w    = (const float*)d_in[15];
  const float* fc1_b    = (const float*)d_in[16];
  const float* fc2_w    = (const float*)d_in[17];
  const float* fc2_b    = (const float*)d_in[18];
  const float* norm_g   = (const float*)d_in[19];
  const float* norm_b   = (const float*)d_in[20];
  const float* head_w   = (const float*)d_in[21];
  const float* head_b   = (const float*)d_in[22];
  float* out = (float*)d_out;

  // ---- workspace carve-up (byte-based, 256B aligned slabs) ----
  char* p = (char*)d_ws;
  auto alloc = [&](size_t bytes) -> char* {
    char* r = p;
    p += (bytes + 255) & ~(size_t)255;
    return r;
  };
  const size_t szTokF = (size_t)MTOK * Dm * sizeof(float);
  float* bufH0   = (float*)alloc(szTokF);                       // h ping
  float* bufH1   = (float*)alloc(szTokF);                       // h pong
  float* bufX1   = (float*)alloc(szTokF);                       // post-attn residual
  float* bufXe   = (float*)alloc(szTokF);                       // expert output
  float* bufTmpF = (float*)alloc(szTokF);                       // patch GEMM out / final LN
  float* bufRW   = (float*)alloc((size_t)MTOK * Ee * sizeof(float));
  u16* bufLn   = (u16*)alloc((size_t)MTOK * Dm * 2);            // LN out (bf16)
  u16* bufAtt  = (u16*)alloc((size_t)MTOK * Dm * 2);            // attn out (bf16)
  u16* bufQkv  = (u16*)alloc((size_t)MTOK * 768 * 2);           // qkv bf16; im2col reuses
  u16* bufFF   = (u16*)alloc((size_t)MTOK * FFf * 2);           // fc1 out (bf16)
  u16* wPatchT = (u16*)alloc((size_t)Dm * 768 * 2);             // 256 x 768
  u16* wQkvT   = (u16*)alloc((size_t)Ll * Ee * 768 * Dm * 2);   // per le: 768 x 256
  u16* wProjT  = (u16*)alloc((size_t)Ll * Ee * Dm * Dm * 2);    // per le: 256 x 256
  u16* wFc1T   = (u16*)alloc((size_t)Ll * Ee * FFf * Dm * 2);   // per le: 1024 x 256
  u16* wFc2T   = (u16*)alloc((size_t)Ll * Ee * Dm * FFf * 2);   // per le: 256 x 1024

  const dim3 blk256(256);
  auto g1 = [](size_t n) { return dim3((unsigned)((n + 255) / 256)); };

  // ---- 0) convert + transpose all weights to bf16 W^T ----
  k_cvtT<<<g1((size_t)768 * Dm), blk256, 0, stream>>>(patch_w, wPatchT, 768, Dm, 1);
  k_cvtT<<<g1((size_t)Ll * Ee * Dm * 768), blk256, 0, stream>>>(qkv_w, wQkvT, Dm, 768, Ll * Ee);
  k_cvtT<<<g1((size_t)Ll * Ee * Dm * Dm), blk256, 0, stream>>>(proj_w, wProjT, Dm, Dm, Ll * Ee);
  k_cvtT<<<g1((size_t)Ll * Ee * Dm * FFf), blk256, 0, stream>>>(fc1_w, wFc1T, Dm, FFf, Ll * Ee);
  k_cvtT<<<g1((size_t)Ll * Ee * FFf * Dm), blk256, 0, stream>>>(fc2_w, wFc2T, FFf, Dm, Ll * Ee);

  // ---- 1) patch embedding ----
  k_im2col<<<g1((size_t)MPAT * 768), blk256, 0, stream>>>(x, bufQkv);
  k_gemm<0, 0><<<dim3(MPAT / 128, Dm / 64), blk256, 0, stream>>>(
      bufQkv, wPatchT, patch_b, nullptr, bufTmpF, MPAT, Dm, 768);
  k_embed<<<g1((size_t)MTOK * Dm), blk256, 0, stream>>>(bufTmpF, cls_tok, pos_emb, bufH0);

  float* cur = bufH0;
  float* nxt = bufH1;
  const size_t nTok = (size_t)MTOK * Dm;

  for (int l = 0; l < Ll; ++l) {
    k_router<<<g1(MTOK), blk256, 0, stream>>>(
        cur, router_w + (size_t)l * Dm * Ee, router_b + (size_t)l * Ee, bufRW, MTOK);
    k_zero<<<g1(nTok), blk256, 0, stream>>>(nxt, nTok);

    for (int e = 0; e < Ee; ++e) {
      const size_t le = (size_t)l * Ee + e;
      k_ln<1><<<dim3(MTOK / 8), blk256, 0, stream>>>(
          cur, ln1_g + le * Dm, ln1_b + le * Dm, bufLn, MTOK);
      k_gemm<0, 1><<<dim3(MTOK / 128, (3 * Dm) / 64), blk256, 0, stream>>>(
          bufLn, wQkvT + le * 768 * Dm, qkv_b + le * 3 * Dm, nullptr,
          bufQkv, MTOK, 3 * Dm, Dm);
      k_attn<<<dim3(SP / 16, NHh, Bb), dim3(32), 0, stream>>>(bufQkv, bufAtt);
      k_gemm<0, 0><<<dim3(MTOK / 128, Dm / 64), blk256, 0, stream>>>(
          bufAtt, wProjT + le * Dm * Dm, proj_b + le * Dm, cur,
          bufX1, MTOK, Dm, Dm);
      k_ln<1><<<dim3(MTOK / 8), blk256, 0, stream>>>(
          bufX1, ln2_g + le * Dm, ln2_b + le * Dm, bufLn, MTOK);
      k_gemm<1, 1><<<dim3(MTOK / 128, FFf / 64), blk256, 0, stream>>>(
          bufLn, wFc1T + le * FFf * Dm, fc1_b + le * FFf, nullptr,
          bufFF, MTOK, FFf, Dm);
      k_gemm<0, 0><<<dim3(MTOK / 128, Dm / 64), blk256, 0, stream>>>(
          bufFF, wFc2T + le * Dm * FFf, fc2_b + le * Dm, bufX1,
          bufXe, MTOK, Dm, FFf);
      k_accum<<<g1(nTok), blk256, 0, stream>>>(nxt, bufXe, bufRW, e, nTok);
    }
    float* t = cur; cur = nxt; nxt = t;
  }

  // ---- final LN + classifier head on cls tokens ----
  k_ln<0><<<dim3(MTOK / 8), blk256, 0, stream>>>(cur, norm_g, norm_b, bufTmpF, MTOK);
  k_head<<<g1(Bb * NCc), blk256, 0, stream>>>(bufTmpF, head_w, head_b, out);
}
